// SelfAttention_12652973654654
// MI455X (gfx1250) — matrix operand order repaired and checked by reading, not compiled, not run on hardware
//
#include <hip/hip_runtime.h>
#include <hip/hip_bf16.h>
#include <math.h>

#define S_LEN 4096
#define HIDDEN 2048
#define NH 16
#define NKV 4
#define HD 128
#define NQKV 3072  // 2048 q + 512 k + 512 v

typedef __bf16 bf16_t;
typedef __attribute__((ext_vector_type(16))) __bf16 v16bf;
typedef __attribute__((ext_vector_type(8)))  __bf16 v8bf;
typedef __attribute__((ext_vector_type(8)))  float  v8f;

static __device__ __forceinline__ float fast_exp2(float x) {
  return __builtin_amdgcn_exp2f(x);  // v_exp_f32
}

static __device__ __forceinline__ v8f zero_v8f() {
  v8f z = {0.f, 0.f, 0.f, 0.f, 0.f, 0.f, 0.f, 0.f};
  return z;
}

// Build a 16-element bf16 fragment from two 16-byte (b128) loads.
static __device__ __forceinline__ v16bf load_frag(const bf16_t* p_lo, const bf16_t* p_hi) {
  v8bf lo = *(const v8bf*)p_lo;
  v8bf hi = *(const v8bf*)p_hi;
  v16bf r;
#pragma unroll
  for (int i = 0; i < 8; ++i) { r[i] = lo[i]; r[i + 8] = hi[i]; }
  return r;
}

// ---------------- prep kernels ----------------

__global__ void cast_f32_bf16(const float* __restrict__ src, bf16_t* __restrict__ dst, int n) {
  int i = blockIdx.x * blockDim.x + threadIdx.x;
  if (i < n) dst[i] = (bf16_t)src[i];
}

// dst[n][k] = src[k][n]  (weights K x N -> transposed bf16 N x K)
__global__ void transpose_cast(const float* __restrict__ src, bf16_t* __restrict__ dst,
                               int K, int N) {
  int i = blockIdx.x * blockDim.x + threadIdx.x;
  if (i >= K * N) return;
  int n = i % N;
  int k = i / N;
  dst[(size_t)n * K + k] = (bf16_t)src[(size_t)k * N + n];
}

// ---------------- bf16 WMMA GEMM:  C[M][N] = A[M][K] * Bt[N][K]^T ----------------
// 256 threads = 8 waves; each wave computes a 32x32 tile (2x2 WMMA tiles),
// block covers 64 (M) x 128 (N).
template <typename OutT>
__global__ void gemm_bf16_wmma(const bf16_t* __restrict__ A, const bf16_t* __restrict__ Bt,
                               OutT* __restrict__ C, int M, int N, int K) {
  const int wave = threadIdx.x >> 5;
  const int lane = threadIdx.x & 31;
  const int half = lane >> 4;
  const int l15  = lane & 15;
  const int wm = wave >> 2;  // 0..1
  const int wn = wave & 3;   // 0..3
  const int mbase = blockIdx.y * 64 + wm * 32;
  const int nbase = blockIdx.x * 128 + wn * 32;

  v8f c00 = zero_v8f(), c01 = zero_v8f(), c10 = zero_v8f(), c11 = zero_v8f();
  const int koffA = half * 8;   // A frag: elems 0..7 = k+0..7, elems 8..15 = k+16..23
  const int koffB = half * 8;   // B frag: elems 0..7 = k+0..7, elems 8..15 = k+16..23
  const size_t rowA0 = (size_t)(mbase + l15) * K;
  const size_t rowA1 = (size_t)(mbase + 16 + l15) * K;
  const size_t rowB0 = (size_t)(nbase + l15) * K;
  const size_t rowB1 = (size_t)(nbase + 16 + l15) * K;

  for (int kk = 0; kk < K; kk += 32) {
    v16bf a0 = load_frag(A + rowA0 + kk + koffA, A + rowA0 + kk + koffA + 16);
    v16bf a1 = load_frag(A + rowA1 + kk + koffA, A + rowA1 + kk + koffA + 16);
    v16bf b0 = load_frag(Bt + rowB0 + kk + koffB, Bt + rowB0 + kk + koffB + 16);
    v16bf b1 = load_frag(Bt + rowB1 + kk + koffB, Bt + rowB1 + kk + koffB + 16);
    c00 = __builtin_amdgcn_wmma_f32_16x16x32_bf16(false, a0, false, b0, (short)0, c00, false, false);
    c01 = __builtin_amdgcn_wmma_f32_16x16x32_bf16(false, a0, false, b1, (short)0, c01, false, false);
    c10 = __builtin_amdgcn_wmma_f32_16x16x32_bf16(false, a1, false, b0, (short)0, c10, false, false);
    c11 = __builtin_amdgcn_wmma_f32_16x16x32_bf16(false, a1, false, b1, (short)0, c11, false, false);
  }

#pragma unroll
  for (int e = 0; e < 8; ++e) {
    int r0 = mbase + e + half * 8;
    int r1 = r0 + 16;
    int n0 = nbase + l15;
    int n1 = n0 + 16;
    C[(size_t)r0 * N + n0] = (OutT)c00[e];
    C[(size_t)r0 * N + n1] = (OutT)c01[e];
    C[(size_t)r1 * N + n0] = (OutT)c10[e];
    C[(size_t)r1 * N + n1] = (OutT)c11[e];
  }
}

// ---------------- RoPE + layout kernels ----------------

__global__ void rope_q_kernel(const bf16_t* __restrict__ qkv, const int* __restrict__ pos_ids,
                              bf16_t* __restrict__ Qr) {
  int idx = blockIdx.x * blockDim.x + threadIdx.x;
  if (idx >= NH * 64 * S_LEN) return;
  int di = idx % 64;
  int h  = (idx / 64) % NH;
  int s  = idx / (64 * NH);
  size_t src = (size_t)s * NQKV + h * HD + di;
  float x1 = (float)qkv[src];
  float x2 = (float)qkv[src + 64];
  float pos = (float)pos_ids[s];
  float freq = __powf(10000.0f, -(float)(2 * di) * (1.0f / 128.0f));
  float sn, cs;
  __sincosf(pos * freq, &sn, &cs);
  size_t dst = ((size_t)h * S_LEN + s) * HD + di;
  Qr[dst]      = (bf16_t)(x1 * cs - x2 * sn);
  Qr[dst + 64] = (bf16_t)(x2 * cs + x1 * sn);
}

__global__ void rope_k_kernel(const bf16_t* __restrict__ qkv, const int* __restrict__ pos_ids,
                              bf16_t* __restrict__ Kr) {
  int idx = blockIdx.x * blockDim.x + threadIdx.x;
  if (idx >= NKV * 64 * S_LEN) return;
  int di = idx % 64;
  int kv = (idx / 64) % NKV;
  int s  = idx / (64 * NKV);
  size_t src = (size_t)s * NQKV + HIDDEN + kv * HD + di;
  float x1 = (float)qkv[src];
  float x2 = (float)qkv[src + 64];
  float pos = (float)pos_ids[s];
  float freq = __powf(10000.0f, -(float)(2 * di) * (1.0f / 128.0f));
  float sn, cs;
  __sincosf(pos * freq, &sn, &cs);
  size_t dst = ((size_t)kv * S_LEN + s) * HD + di;
  Kr[dst]      = (bf16_t)(x1 * cs - x2 * sn);
  Kr[dst + 64] = (bf16_t)(x2 * cs + x1 * sn);
}

// Vt[kv][d][s] = V[s][kv][d]  (transposed so V is a valid WMMA B-operand for P*V)
__global__ void v_transpose_kernel(const bf16_t* __restrict__ qkv, bf16_t* __restrict__ Vt) {
  int idx = blockIdx.x * blockDim.x + threadIdx.x;
  if (idx >= NKV * HD * S_LEN) return;
  int s  = idx % S_LEN;
  int d  = (idx / S_LEN) % HD;
  int kv = idx / (S_LEN * HD);
  Vt[((size_t)kv * HD + d) * S_LEN + s] =
      qkv[(size_t)s * NQKV + (HIDDEN + NKV * HD) + kv * HD + d];
}

// ---------------- flash attention (wave-level, causal, GQA) ----------------
// 128 threads = 4 waves = exactly 1 wave per SIMD32 -> full VGPR budget per
// wave (no spills), and back-to-back WMMA issue is cheapest in this regime.
// Each wave: 16 query rows, 16x128 fp32 accumulator resident in VGPRs.
// Key blocks of 32: 8 score WMMAs + online softmax + LDS re-layout + 8 PV WMMAs.
__global__ void __launch_bounds__(128, 1)
flash_attn_kernel(const bf16_t* __restrict__ Qr, const bf16_t* __restrict__ Kr,
                  const bf16_t* __restrict__ Vt, const int* __restrict__ amask,
                  bf16_t* __restrict__ Obf) {
  __shared__ __align__(16) bf16_t pbuf[4][16 * 32];  // per-wave P tile scratch
  const int wave = threadIdx.x >> 5;
  const int lane = threadIdx.x & 31;
  const int half = lane >> 4;
  const int l15  = lane & 15;
  const int head = blockIdx.y;
  const int kv   = head >> 2;  // GQA: 4 q-heads per kv-head
  const int qbase = blockIdx.x * 64 + wave * 16;

  const bf16_t* Qh = Qr + (size_t)head * S_LEN * HD;
  const bf16_t* Kh = Kr + (size_t)kv * S_LEN * HD;
  const bf16_t* Vh = Vt + (size_t)kv * HD * S_LEN;

  const int koffA = half * 8;
  const int koffB = half * 8;

  // Q tile as 4 A-fragments (K = 128 split into 4x32)
  v16bf qf[4];
  {
    size_t qrow = (size_t)(qbase + l15) * HD;
#pragma unroll
    for (int dc = 0; dc < 4; ++dc) {
      size_t b = qrow + dc * 32 + koffA;
      qf[dc] = load_frag(Qh + b, Qh + b + 16);
    }
  }

  v8f acc[8];
#pragma unroll
  for (int nt = 0; nt < 8; ++nt) acc[nt] = zero_v8f();
  float m_run[8], l_run[8];
#pragma unroll
  for (int e = 0; e < 8; ++e) { m_run[e] = -1e30f; l_run[e] = 0.0f; }

  // (1/sqrt(128)) * log2(e): softmax tracked in exp2 domain
  const float SCL = 0.08838834764831845f * 1.4426950408889634f;

  const int nkb = (qbase + 47) >> 5;  // causal: keys 0 .. qbase+15
  for (int kb = 0; kb < nkb; ++kb) {
    const int kbase = kb * 32;

    // scores: two 16x16 tiles (32 keys), each = 4 WMMAs over d
    v8f sc[2];
#pragma unroll
    for (int t = 0; t < 2; ++t) {
      v8f c = zero_v8f();
      size_t krow = (size_t)(kbase + t * 16 + l15) * HD;
#pragma unroll
      for (int dc = 0; dc < 4; ++dc) {
        size_t b = krow + dc * 32 + koffB;
        v16bf kf = load_frag(Kh + b, Kh + b + 16);
        c = __builtin_amdgcn_wmma_f32_16x16x32_bf16(false, qf[dc], false, kf, (short)0, c, false, false);
      }
      sc[t] = c;
    }

    const int key0 = kbase + l15;
    const int key1 = key0 + 16;
    const int am0 = amask[key0];
    const int am1 = amask[key1];

#pragma unroll
    for (int e = 0; e < 8; ++e) {
      int qr = qbase + e + half * 8;
      float u0 = sc[0][e] * SCL;
      float u1 = sc[1][e] * SCL;
      if (key0 > qr || am0 == 0) u0 = -1e30f;
      if (key1 > qr || am1 == 0) u1 = -1e30f;
      float rm = fmaxf(u0, u1);
#pragma unroll
      for (int off = 8; off >= 1; off >>= 1)
        rm = fmaxf(rm, __shfl_xor(rm, off, 32));
      float m_new = fmaxf(m_run[e], rm);
      float alpha = fast_exp2(m_run[e] - m_new);
      float p0 = fast_exp2(u0 - m_new);
      float p1 = fast_exp2(u1 - m_new);
      float ps = p0 + p1;
#pragma unroll
      for (int off = 8; off >= 1; off >>= 1)
        ps += __shfl_xor(ps, off, 32);
      l_run[e] = l_run[e] * alpha + ps;
      m_run[e] = m_new;
#pragma unroll
      for (int nt = 0; nt < 8; ++nt) acc[nt][e] *= alpha;
      // C-layout -> LDS (row-major 16x32) for the A-layout reload
      int row = e + half * 8;
      pbuf[wave][row * 32 + l15]      = (bf16_t)p0;
      pbuf[wave][row * 32 + 16 + l15] = (bf16_t)p1;
    }

    // reload P as A-fragment (DS ops are in-order per wave on CDNA5)
    {
      int prow = l15 * 32 + koffA;
      v16bf pf = load_frag(&pbuf[wave][prow], &pbuf[wave][prow + 16]);
#pragma unroll
      for (int nt = 0; nt < 8; ++nt) {
        size_t vrow = (size_t)(nt * 16 + l15) * S_LEN + kbase + koffB;
        v16bf vf = load_frag(Vh + vrow, Vh + vrow + 16);
        acc[nt] = __builtin_amdgcn_wmma_f32_16x16x32_bf16(false, pf, false, vf, (short)0, acc[nt], false, false);
      }
    }
  }

  // normalize and emit bf16 rows for the output projection: Obf[s][head*128 + d]
#pragma unroll
  for (int e = 0; e < 8; ++e) {
    float inv = 1.0f / l_run[e];
    int q = qbase + e + half * 8;
    size_t rowbase = (size_t)q * (NH * HD) + (size_t)head * HD;
#pragma unroll
    for (int nt = 0; nt < 8; ++nt) {
      Obf[rowbase + nt * 16 + l15] = (bf16_t)(acc[nt][e] * inv);
    }
  }
}

// ---------------- launch ----------------

extern "C" void kernel_launch(void* const* d_in, const int* in_sizes, int n_in,
                              void* d_out, int out_size, void* d_ws, size_t ws_size,
                              hipStream_t stream) {
  (void)in_sizes; (void)n_in; (void)out_size; (void)ws_size;
  const float* hs    = (const float*)d_in[0];
  const int*   amask = (const int*)d_in[1];
  const int*   pos   = (const int*)d_in[2];
  const float* wq    = (const float*)d_in[3];
  const float* wk    = (const float*)d_in[4];
  const float* wv    = (const float*)d_in[5];
  const float* wo    = (const float*)d_in[6];
  float* out = (float*)d_out;

  char* ws = (char*)d_ws;
  size_t off = 0;
  bf16_t* Xbf   = (bf16_t*)(ws + off); off += (size_t)S_LEN * HIDDEN * 2;   // 16 MB
  bf16_t* WqkvT = (bf16_t*)(ws + off); off += (size_t)NQKV * HIDDEN * 2;    // 12 MB
  bf16_t* WoT   = (bf16_t*)(ws + off); off += (size_t)HIDDEN * HIDDEN * 2;  //  8 MB
  bf16_t* QKV   = (bf16_t*)(ws + off); off += (size_t)S_LEN * NQKV * 2;     // 24 MB
  bf16_t* Qr    = (bf16_t*)(ws + off); off += (size_t)NH * S_LEN * HD * 2;  // 16 MB
  bf16_t* Kr    = (bf16_t*)(ws + off); off += (size_t)NKV * S_LEN * HD * 2; //  4 MB
  bf16_t* Vt    = (bf16_t*)(ws + off); off += (size_t)NKV * HD * S_LEN * 2; //  4 MB
  bf16_t* Obf   = QKV;  // reuse: QKV consumed before attention output is written

  // 1) casts / weight transposes
  {
    int n = S_LEN * HIDDEN;
    cast_f32_bf16<<<(n + 255) / 256, 256, 0, stream>>>(hs, Xbf, n);
  }
  transpose_cast<<<(HIDDEN * 2048 + 255) / 256, 256, 0, stream>>>(wq, WqkvT, HIDDEN, 2048);
  transpose_cast<<<(HIDDEN * 512 + 255) / 256, 256, 0, stream>>>(
      wk, WqkvT + (size_t)2048 * HIDDEN, HIDDEN, 512);
  transpose_cast<<<(HIDDEN * 512 + 255) / 256, 256, 0, stream>>>(
      wv, WqkvT + (size_t)2560 * HIDDEN, HIDDEN, 512);
  transpose_cast<<<(HIDDEN * HIDDEN + 255) / 256, 256, 0, stream>>>(wo, WoT, HIDDEN, HIDDEN);

  // 2) fused QKV projection: [4096,2048] x [2048,3072]
  gemm_bf16_wmma<bf16_t><<<dim3(NQKV / 128, S_LEN / 64), 256, 0, stream>>>(
      Xbf, WqkvT, QKV, S_LEN, NQKV, HIDDEN);

  // 3) RoPE + attention-friendly layouts
  {
    int nq = NH * 64 * S_LEN;
    rope_q_kernel<<<(nq + 255) / 256, 256, 0, stream>>>(QKV, pos, Qr);
    int nk = NKV * 64 * S_LEN;
    rope_k_kernel<<<(nk + 255) / 256, 256, 0, stream>>>(QKV, pos, Kr);
    int nv = NKV * HD * S_LEN;
    v_transpose_kernel<<<(nv + 255) / 256, 256, 0, stream>>>(QKV, Vt);
  }

  // 4) causal flash attention, one wave per 16-query tile, 1 wave per SIMD
  flash_attn_kernel<<<dim3(S_LEN / 64, NH), 128, 0, stream>>>(Qr, Kr, Vt, amask, Obf);

  // 5) output projection -> fp32 result
  gemm_bf16_wmma<float><<<dim3(HIDDEN / 128, S_LEN / 64), 256, 0, stream>>>(
      Obf, WoT, out, S_LEN, HIDDEN, HIDDEN);
}
